// MyModel_61933428413973
// MI455X (gfx1250) — 8-bit matrix operand order repaired and checked by reading, compiled, not run on hardware
//
#include <hip/hip_runtime.h>


typedef __attribute__((ext_vector_type(16))) int   v16i;
typedef __attribute__((ext_vector_type(8)))  float v8f;
typedef __attribute__((ext_vector_type(4)))  float v4f;

#define MDIM 32768
#define KDIM 1024
#define NDIM 4096
#define KT   (KDIM / 128)
#define FRAG_BYTES 2048

static_assert(MDIM % 256 == 0 && NDIM % 256 == 0 && KDIM % 128 == 0, "tiling");

__device__ __forceinline__ unsigned int f32_to_e4m3(float f) {
  unsigned int u = __float_as_uint(f);
  unsigned int s = (u >> 24) & 0x80u;
  unsigned int a = u & 0x7FFFFFFFu;
  if (a >= 0x7F800000u) return s | 0x7Fu;
  if (a < 0x3C800000u) {
    float af = __uint_as_float(a);
    unsigned int q = (unsigned int)__float2int_rn(af * 512.0f);
    return s | q;
  }
  a += 0x0007FFFFu + ((a >> 20) & 1u);
  int e = (int)(a >> 23) - 127;
  unsigned int frac = (a >> 20) & 7u;
  if (e > 8 || (e == 8 && frac == 7u)) return s | 0x7Eu;
  return s | ((unsigned int)(e + 7) << 3) | frac;
}

__global__ void __launch_bounds__(256) pack_a_e4m3(const float* __restrict__ x,
                                                   unsigned char* __restrict__ ap) {
  unsigned int gid = blockIdx.x * 256u + threadIdx.x;
  unsigned int m   = gid >> 8;
  unsigned int k   = (gid & 255u) << 2;

  v4f xv = *(const v4f*)(x + (size_t)m * KDIM + k);
  unsigned int b = f32_to_e4m3(xv.x) | (f32_to_e4m3(xv.y) << 8) |
                   (f32_to_e4m3(xv.z) << 16) | (f32_to_e4m3(xv.w) << 24);

  unsigned int mt = m >> 4, lrow = m & 15u;
  unsigned int kt = k >> 7, kin = k & 127u;
  unsigned int half = kin >> 6, k64 = kin & 63u;
  unsigned int lhi = (k64 >> 3) & 1u;
  unsigned int lane = lhi * 16u + lrow;
  unsigned int v = ((k64 >> 4) << 1) | ((k64 >> 2) & 1u);
  unsigned int j = half * 32u + v * 4u;
  size_t addr = (size_t)(mt * KT + kt) * FRAG_BYTES + lane * 64u + j;
  *(unsigned int*)(ap + addr) = b;
}

__global__ void __launch_bounds__(256) pack_b_e4m3(const float* __restrict__ w,
                                                   unsigned char* __restrict__ bp) {
  unsigned int gid = blockIdx.x * 256u + threadIdx.x;
  unsigned int n   = gid & (NDIM - 1u);
  unsigned int k   = (gid >> 12) << 2;

  float f0 = w[(size_t)(k + 0) * NDIM + n];
  float f1 = w[(size_t)(k + 1) * NDIM + n];
  float f2 = w[(size_t)(k + 2) * NDIM + n];
  float f3 = w[(size_t)(k + 3) * NDIM + n];
  unsigned int b = f32_to_e4m3(f0) | (f32_to_e4m3(f1) << 8) |
                   (f32_to_e4m3(f2) << 16) | (f32_to_e4m3(f3) << 24);

  unsigned int nt = n >> 4, lrow = n & 15u;
  unsigned int kt = k >> 7, kin = k & 127u;
  unsigned int lhi = (kin >> 3) & 1u;
  unsigned int lane = lhi * 16u + lrow;
  unsigned int j = ((kin >> 6) << 5) | (((kin >> 4) & 3u) << 3) | (kin & 7u);
  size_t addr = (size_t)(nt * KT + kt) * FRAG_BYTES + lane * 64u + j;
  *(unsigned int*)(bp + addr) = b;
}

__global__ void __launch_bounds__(256) gemm_fp8_wmma(const unsigned char* __restrict__ ap,
                                                     const unsigned char* __restrict__ bp,
                                                     float* __restrict__ c) {
  const unsigned int lane = threadIdx.x & 31u;
  const unsigned int wave = threadIdx.x >> 5u;
  const unsigned int bm   = blockIdx.x >> 4u;
  const unsigned int bn   = blockIdx.x & 15u;
  const unsigned int mt128 = bm * 2u + (wave >> 2u);
  const unsigned int nt64  = bn * 4u + (wave & 3u);

  const unsigned char* abase = ap + (size_t)(mt128 * 8u) * KT * FRAG_BYTES + lane * 64u;
  const unsigned char* bbase = bp + (size_t)(nt64 * 4u) * KT * FRAG_BYTES + lane * 64u;

  v8f acc[8][4];
#pragma unroll
  for (int i = 0; i < 8; ++i)
#pragma unroll
    for (int j = 0; j < 4; ++j) acc[i][j] = (v8f){};

  for (unsigned int kt = 0; kt < KT; ++kt) {
    v16i b[4];
#pragma unroll
    for (int j = 0; j < 4; ++j)
      b[j] = *(const v16i*)(bbase + (size_t)(j * KT + (int)kt) * FRAG_BYTES);
#pragma unroll
    for (int i = 0; i < 8; ++i) {
      v16i a = *(const v16i*)(abase + (size_t)(i * KT + (int)kt) * FRAG_BYTES);
#pragma unroll
      for (int j = 0; j < 4; ++j)
        acc[i][j] = __builtin_amdgcn_wmma_f32_16x16x128_fp8_fp8(
            a, b[j], (short)0, acc[i][j], false, false);
    }
  }

  const unsigned int row0 = mt128 * 128u + (lane >> 4u) * 8u;
  const unsigned int col0 = nt64 * 64u + (lane & 15u);
#pragma unroll
  for (int i = 0; i < 8; ++i)
#pragma unroll
    for (int j = 0; j < 4; ++j)
#pragma unroll
      for (int r = 0; r < 8; ++r)
        __builtin_nontemporal_store(
            acc[i][j][r],
            c + (size_t)(row0 + i * 16u + r) * NDIM + col0 + j * 16u);
}

extern "C" void kernel_launch(void* const* d_in, const int* in_sizes, int n_in,
                              void* d_out, int out_size, void* d_ws, size_t ws_size,
                              hipStream_t stream) {
  (void)in_sizes; (void)n_in; (void)out_size; (void)ws_size;
  const float* x = (const float*)d_in[0];
  const float* w = (const float*)d_in[1];
  float* c = (float*)d_out;

  unsigned char* ap = (unsigned char*)d_ws;
  unsigned char* bp = ap + (size_t)MDIM * KDIM;

  pack_a_e4m3<<<(MDIM * (KDIM / 4)) / 256, 256, 0, stream>>>(x, ap);
  pack_b_e4m3<<<((KDIM / 4) * NDIM) / 256, 256, 0, stream>>>(w, bp);

  const unsigned int waves = (MDIM / 128) * (NDIM / 64);
  gemm_fp8_wmma<<<waves / 8, 256, 0, stream>>>(ap, bp, c);
}
